// DiffusionNet_occ_21603685499588
// MI455X (gfx1250) — matrix operand order repaired and checked by reading, not compiled, not run on hardware
//
#include <hip/hip_runtime.h>
#include <math.h>

// ---------------------------------------------------------------------------
// DiffusionNet forward for MI455X (gfx1250, wave32).
// GEMMs: v_wmma_f32_16x16x32_bf16, bf16 operands staged to LDS with
// GLOBAL_LOAD_ASYNC_TO_LDS_B128 (double-buffered, ASYNCcnt-tracked), fp32
// accumulate. All operand tensors are pre-materialized in bf16 with k-major
// contiguous layout and padded shapes so the staging loop is branch-free.
// ---------------------------------------------------------------------------

typedef __attribute__((ext_vector_type(16))) __bf16        v16bf;
typedef __attribute__((ext_vector_type(8)))  float         v8f;
typedef __attribute__((ext_vector_type(4)))  unsigned int  v4u;
typedef int v4i __attribute__((vector_size(16)));            // builtin's vector type
typedef __attribute__((address_space(1))) v4i* gp4;          // global int4*
typedef __attribute__((address_space(3))) v4i* lp4;          // LDS int4*

#define B_    4
#define N_    1024
#define H_    8
#define NP1_  1025
#define KPAD_ 1056     // 1025 padded to multiple of 32
#define NKPAD_ 1088    // 1025 key rows padded to multiple of 64
#define INNER_ 512

#if defined(__has_builtin)
#  if __has_builtin(__builtin_amdgcn_global_load_async_to_lds_b128)
#    define USE_ASYNC 1
#  endif
#endif
#ifndef USE_ASYNC
#  define USE_ASYNC 0
#endif

__device__ __forceinline__ unsigned short bf16b(float f) {
  unsigned int u = __float_as_uint(f);
  return (unsigned short)((u + 0x7FFFu + ((u >> 16) & 1u)) >> 16);   // RNE
}
__device__ __forceinline__ float bf2f(unsigned short s) {
  return __uint_as_float((unsigned int)s << 16);
}

__device__ __forceinline__ void wait_async0() {
#if USE_ASYNC
#  if __has_builtin(__builtin_amdgcn_s_wait_asynccnt)
  __builtin_amdgcn_s_wait_asynccnt(0);
#  else
  asm volatile("s_wait_asynccnt 0x0" ::: "memory");
#  endif
#endif
}

// ---------------------------------------------------------------------------
// WMMA GEMM:  C[z] = A[z] x B[z/divB]^T-ish
//   A bf16 [M,K]  row-major (lda), M % 64 == 0, K % 32 == 0
//   B bf16 [N,K]  row-major (ldb)  (i.e. n-th row = column n of the math B)
//   C fp32 [M,N]  (ldc), store guarded by n < N
// Block 128 thr = 4 waves; block tile 64x64; wave tile 16x64; BK=32.
// LDS double-buffered; tiles streamed with async-to-LDS when available.
// ---------------------------------------------------------------------------
__global__ __launch_bounds__(128)
void k_gemm_bf16(const unsigned short* __restrict__ A,
                 const unsigned short* __restrict__ Bm,
                 float* __restrict__ C, int M, int N, int K,
                 int lda, int ldb, int ldc,
                 long long sA, long long sB, long long sC, int divB)
{
  A  += (long long)blockIdx.z * sA;
  Bm += (long long)(blockIdx.z / divB) * sB;
  C  += (long long)blockIdx.z * sC;

  const int bm0 = blockIdx.y * 64;
  const int bn0 = blockIdx.x * 64;

  __shared__ alignas(16) unsigned short As[2][64 * 32];   // [m][k]
  __shared__ alignas(16) unsigned short Bs[2][64 * 32];   // [n][k]

  const int tid  = threadIdx.x;
  const int wave = tid >> 5;
  const int lane = tid & 31;

  const unsigned short* Arow = A  + (long long)bm0 * lda;
  const unsigned short* Brow = Bm + (long long)bn0 * ldb;

  // each thread stages 2x16B for A and 2x16B for B per tile (branch-free)
  const int rr = tid >> 2;            // 0..31
  const int oo = (tid & 3) * 8;       // ushort offset: 0,8,16,24

  v8f zero8 = {0.f,0.f,0.f,0.f,0.f,0.f,0.f,0.f};
  v8f acc[4];
  acc[0] = zero8; acc[1] = zero8; acc[2] = zero8; acc[3] = zero8;

  const int ktiles = K >> 5;

#if USE_ASYNC
  auto issue = [&](int kt, int pb) {
    const int k0 = kt << 5;
#pragma unroll
    for (int half = 0; half < 2; ++half) {
      const int r = rr + half * 32;
      __builtin_amdgcn_global_load_async_to_lds_b128(
          (gp4)(Arow + (long long)r * lda + k0 + oo),
          (lp4)&As[pb][r * 32 + oo], 0, 0);
      __builtin_amdgcn_global_load_async_to_lds_b128(
          (gp4)(Brow + (long long)r * ldb + k0 + oo),
          (lp4)&Bs[pb][r * 32 + oo], 0, 0);
    }
  };
  issue(0, 0);
  int p = 0;
#endif

  for (int kt = 0; kt < ktiles; ++kt) {
#if USE_ASYNC
    wait_async0();
    __syncthreads();
    if (kt + 1 < ktiles) issue(kt + 1, p ^ 1);
    const int pb = p;
    p ^= 1;
#else
    const int k0 = kt << 5;
    __syncthreads();
#pragma unroll
    for (int half = 0; half < 2; ++half) {
      const int r = rr + half * 32;
      *(v4u*)&As[0][r * 32 + oo] = *(const v4u*)(Arow + (long long)r * lda + k0 + oo);
      *(v4u*)&Bs[0][r * 32 + oo] = *(const v4u*)(Brow + (long long)r * ldb + k0 + oo);
    }
    __syncthreads();
    const int pb = 0;
#endif

    union Frag { v4u q[2]; v16bf v; };
    Frag fa;
    const int mm  = wave * 16 + (lane & 15);
    const int kbA = (lane >> 4) * 8;           // A: lanes16-31 hold K=8..15 / 24..31
    fa.q[0] = *(const v4u*)&As[pb][mm * 32 + kbA];
    fa.q[1] = *(const v4u*)&As[pb][mm * 32 + 16 + kbA];
    const int kbB = (lane >> 4) * 8;           // B: lanes16-31 hold K=8..15 / 24..31
#pragma unroll
    for (int j = 0; j < 4; ++j) {
      Frag fb;
      const int nn = j * 16 + (lane & 15);
      fb.q[0] = *(const v4u*)&Bs[pb][nn * 32 + kbB];
      fb.q[1] = *(const v4u*)&Bs[pb][nn * 32 + 16 + kbB];
      acc[j] = __builtin_amdgcn_wmma_f32_16x16x32_bf16(
                 false, fa.v, false, fb.v, (short)0, acc[j], false, false);
    }
  }

  const int mbase = bm0 + wave * 16 + (lane >> 4) * 8;
  const int nb    = bn0 + (lane & 15);
#pragma unroll
  for (int j = 0; j < 4; ++j) {
    const int n = nb + j * 16;
    if (n < N) {
#pragma unroll
      for (int i = 0; i < 8; ++i)
        C[(long long)(mbase + i) * ldc + n] = acc[j][i];
    }
  }
}

// ---------------------------------------------------------------------------
// LayerNorm (optionally "stable"). OUT = float or bf16(ushort). Block per row.
// ---------------------------------------------------------------------------
__device__ __forceinline__ void stval(float* p, float v)          { *p = v; }
__device__ __forceinline__ void stval(unsigned short* p, float v) { *p = bf16b(v); }

template <typename OUT>
__global__ __launch_bounds__(256)
void k_ln(const float* __restrict__ x, const float* __restrict__ g,
          OUT* __restrict__ out, int D, int stable)
{
  const long long row = blockIdx.x;
  const float* xr = x + row * (long long)D;
  OUT* orow = out + row * (long long)D;
  __shared__ float red[256];
  const int t = threadIdx.x;

  float scale = 1.f;
  if (stable) {
    float mx = -3.4e38f;
    for (int i = t; i < D; i += 256) mx = fmaxf(mx, xr[i]);
    red[t] = mx; __syncthreads();
    for (int s = 128; s > 0; s >>= 1) { if (t < s) red[t] = fmaxf(red[t], red[t + s]); __syncthreads(); }
    scale = 1.f / red[0]; __syncthreads();
  }
  float sum = 0.f;
  for (int i = t; i < D; i += 256) sum += xr[i] * scale;
  red[t] = sum; __syncthreads();
  for (int s = 128; s > 0; s >>= 1) { if (t < s) red[t] += red[t + s]; __syncthreads(); }
  const float mean = red[0] / (float)D; __syncthreads();
  float vs = 0.f;
  for (int i = t; i < D; i += 256) { float d = xr[i] * scale - mean; vs += d * d; }
  red[t] = vs; __syncthreads();
  for (int s = 128; s > 0; s >>= 1) { if (t < s) red[t] += red[t + s]; __syncthreads(); }
  const float inv = rsqrtf(red[0] / (float)D + 1e-5f); __syncthreads();
  for (int i = t; i < D; i += 256) stval(&orow[i], (xr[i] * scale - mean) * inv * g[i]);
}

// ---------------------------------------------------------------------------
// Softmax: fp32 scores (stride rowpad) -> bf16 probs (stride rowpad, tail 0).
// Optional rel-pos bias[h,i,j] (stride rowlen) and causal mask.
// ---------------------------------------------------------------------------
__global__ __launch_bounds__(128)
void k_softmax(const float* __restrict__ sim, const float* __restrict__ bias,
               unsigned short* __restrict__ out,
               int Nrows, int Hh, int rowlen, int rowpad, int causal)
{
  const long long row = blockIdx.x;
  const int i = (int)(row % Nrows);
  const int h = (int)((row / Nrows) % Hh);
  const float* sr = sim + row * (long long)rowpad;
  unsigned short* orow = out + row * (long long)rowpad;
  const float* br = bias ? bias + ((long long)h * Nrows + i) * rowlen : nullptr;
  const int lim = causal ? (i + (rowlen - Nrows)) : (rowlen - 1);
  __shared__ float red[128];
  const int t = threadIdx.x;

  float mx = -3.4e38f;
  for (int j = t; j < rowlen; j += 128) {
    if (causal && j > lim) continue;
    mx = fmaxf(mx, sr[j] + (br ? br[j] : 0.f));
  }
  red[t] = mx; __syncthreads();
  for (int s = 64; s > 0; s >>= 1) { if (t < s) red[t] = fmaxf(red[t], red[t + s]); __syncthreads(); }
  mx = red[0]; __syncthreads();

  float sum = 0.f;
  for (int j = t; j < rowlen; j += 128) {
    if (causal && j > lim) continue;
    sum += expf(sr[j] + (br ? br[j] : 0.f) - mx);
  }
  red[t] = sum; __syncthreads();
  for (int s = 64; s > 0; s >>= 1) { if (t < s) red[t] += red[t + s]; __syncthreads(); }
  const float inv = 1.f / red[0]; __syncthreads();

  for (int j = t; j < rowpad; j += 128) {
    float o = 0.f;
    if (j < rowlen && !(causal && j > lim))
      o = expf(sr[j] + (br ? br[j] : 0.f) - mx) * inv;
    orow[j] = bf16b(o);
  }
}

// ---------------------------------------------------------------------------
// Rotary in place on bf16 rows of 64 (ROT_DIM=32). One thread per pair.
// ---------------------------------------------------------------------------
__global__ void k_rotary(unsigned short* __restrict__ t, int posmod, int posoff,
                         long long total)
{
  const long long idx = blockIdx.x * 256ll + threadIdx.x;
  if (idx >= total) return;
  const int i = (int)(idx & 15);
  const long long row = idx >> 4;
  const int p = (int)(row % posmod) + posoff;
  if (p < 0) return;
  unsigned short* tr = t + row * 64;
  const float invf = expf(-((float)(2 * i) * (1.f / 32.f)) * 9.210340371976184f);
  const float f = (float)p * invf;
  const float c = cosf(f), s = sinf(f);
  const float x1 = bf2f(tr[i]), x2 = bf2f(tr[i + 16]);
  tr[i]      = bf16b(x1 * c - x2 * s);
  tr[i + 16] = bf16b(x2 * c + x1 * s);
}

// fp32 (b,n,:,[colOff + h*64 + d]) -> bf16 (b,h,n,d) [dT=0] or (b,h,d,n) [dT=1]
__global__ void k_to_bhnd(const float* __restrict__ src, int srcLd, int colOff,
                          unsigned short* __restrict__ dst, float alpha, int dT,
                          long long total)
{
  const long long idx = blockIdx.x * 256ll + threadIdx.x;
  if (idx >= total) return;
  const int d = (int)(idx & 63);
  long long u = idx >> 6;
  const int n = (int)(u % N_); u /= N_;
  const int h = (int)(u % H_);
  const long long b = u / H_;
  const float v = alpha * src[(b * N_ + n) * (long long)srcLd + colOff + h * 64 + d];
  const long long di = dT ? (((b * H_ + h) * 64 + d) * (long long)N_ + n) : idx;
  dst[di] = bf16b(v);
}

// fp32 (b,h,n,d) -> bf16 (b,n,h*d)
__global__ void k_to_bnhd(const float* __restrict__ src, unsigned short* __restrict__ dst,
                          long long total)
{
  const long long idx = blockIdx.x * 256ll + threadIdx.x;
  if (idx >= total) return;
  const int d = (int)(idx & 63);
  long long u = idx >> 6;
  const int n = (int)(u % N_); u /= N_;
  const int h = (int)(u % H_);
  const long long b = u / H_;
  dst[(b * N_ + n) * (long long)INNER_ + h * 64 + d] = bf16b(src[idx]);
}

// keys: bf16 [b, NKPAD_, 64]; row 0 = null_kv, rows 1..1024 = kv, rest 0
__global__ void k_build_k(const float* __restrict__ kv, const float* __restrict__ nkv,
                          unsigned short* __restrict__ kf, long long total)
{
  const long long idx = blockIdx.x * 256ll + threadIdx.x;
  if (idx >= total) return;
  const int d = (int)(idx & 63);
  long long u = idx >> 6;
  const int j = (int)(u % NKPAD_);
  const long long b = u / NKPAD_;
  float v = 0.f;
  if (j == 0) v = nkv[d];
  else if (j <= 1024) v = kv[(b * 1024 + (j - 1)) * 128ll + d];
  kf[idx] = bf16b(v);
}

// values^T: bf16 [b, 64, KPAD_]; col 0 = null_kv, cols 1..1024 = kv, rest 0
__global__ void k_build_vt(const float* __restrict__ kv, const float* __restrict__ nkv,
                           unsigned short* __restrict__ vt, long long total)
{
  const long long idx = blockIdx.x * 256ll + threadIdx.x;
  if (idx >= total) return;
  const int j = (int)(idx % KPAD_);
  long long u = idx / KPAD_;
  const int d = (int)(u % 64);
  const long long b = u / 64;
  float v = 0.f;
  if (j == 0) v = nkv[64 + d];
  else if (j <= 1024) v = kv[(b * 1024 + (j - 1)) * 128ll + 64 + d];
  vt[idx] = bf16b(v);
}

// T5 relative position bias (fp32): bias[h,i,j] = emb[bucket(i,j), h]
__global__ void k_relbias(const float* __restrict__ emb, float* __restrict__ bias)
{
  const long long idx = blockIdx.x * 256ll + threadIdx.x;
  if (idx >= (long long)N_ * NP1_) return;
  const int j = (int)(idx % NP1_);
  const int i = (int)(idx / NP1_);
  int neg = i - j; if (neg < 0) neg = 0;
  int bucket;
  if (neg < 16) bucket = neg;
  else {
    int lg = 16 + (int)(logf((float)neg * (1.f / 16.f)) * (16.f / logf(8.f)));
    bucket = lg < 31 ? lg : 31;
  }
#pragma unroll
  for (int h = 0; h < H_; ++h)
    bias[((long long)h * N_ + i) * NP1_ + j] = emb[bucket * H_ + h];
}

// fp32 h[r, 2*inner] -> bf16 out[r, inner] = a * silu(gate)
__global__ void k_silu(const float* __restrict__ h, unsigned short* __restrict__ out,
                       int inner, long long total)
{
  const long long idx = blockIdx.x * 256ll + threadIdx.x;
  if (idx >= total) return;
  const long long r = idx / inner;
  const int c = (int)(idx % inner);
  const float a = h[r * (2ll * inner) + c];
  const float g = h[r * (2ll * inner) + inner + c];
  out[idx] = bf16b(a * (g / (1.f + expf(-g))));
}

// fp32 [R,C] -> bf16 [C,R] (weights to n-major/k-contiguous layout)
__global__ void k_castT(const float* __restrict__ src, unsigned short* __restrict__ dst,
                        int R, int C, long long total)
{
  const long long idx = blockIdx.x * 256ll + threadIdx.x;
  if (idx >= total) return;
  const long long r = idx / C;
  const int c = (int)(idx % C);
  dst[(long long)c * R + r] = bf16b(src[idx]);
}

__global__ void k_add(float* __restrict__ o, const float* __restrict__ a,
                      const float* __restrict__ b, long long total)
{
  const long long idx = blockIdx.x * 256ll + threadIdx.x;
  if (idx < total) o[idx] = a[idx] + b[idx];
}

__global__ void k_copy(float* __restrict__ o, const float* __restrict__ a, long long total)
{
  const long long idx = blockIdx.x * 256ll + threadIdx.x;
  if (idx < total) o[idx] = a[idx];
}

// ---------------------------------------------------------------------------
// Host orchestration
// ---------------------------------------------------------------------------
static inline unsigned cdiv(long long a, long long b) { return (unsigned)((a + b - 1) / b); }

static void gemm(hipStream_t s, const unsigned short* A, const unsigned short* B, float* C,
                 int M, int N, int K, int lda, int ldb, int ldc,
                 long long sA, long long sB, long long sC, int divB, int batch)
{
  dim3 g(cdiv(N, 64), cdiv(M, 64), batch), blk(128);
  k_gemm_bf16<<<g, blk, 0, s>>>(A, B, C, M, N, K, lda, ldb, ldc, sA, sB, sC, divB);
}

static void castT(hipStream_t s, const float* src, unsigned short* dst, int R, int C)
{
  const long long total = (long long)R * C;
  k_castT<<<cdiv(total, 256), 256, 0, s>>>(src, dst, R, C, total);
}

extern "C" void kernel_launch(void* const* d_in, const int* in_sizes, int n_in,
                              void* d_out, int out_size, void* d_ws, size_t ws_size,
                              hipStream_t stream)
{
  (void)in_sizes; (void)n_in; (void)out_size; (void)ws_size;
  const float* x_in = (const float*)d_in[0];
  const float* ctx  = (const float*)d_in[1];

  struct LP {
    const float *sng,*swq,*swkv,*snkv,*swo,*song;
    const float *cng,*cwq,*cwkv,*cnkv,*cwo,*cong,*cctx;
    const float *hng,*hwqkv,*hwo,*hong;
    const float *fng,*fw1,*fw2;
  } L[6];
  int p = 2;
  for (int i = 0; i < 6; ++i) {
    L[i].sng  = (const float*)d_in[p++]; L[i].swq  = (const float*)d_in[p++];
    L[i].swkv = (const float*)d_in[p++]; L[i].snkv = (const float*)d_in[p++];
    L[i].swo  = (const float*)d_in[p++]; L[i].song = (const float*)d_in[p++];
    L[i].cng  = (const float*)d_in[p++]; L[i].cwq  = (const float*)d_in[p++];
    L[i].cwkv = (const float*)d_in[p++]; L[i].cnkv = (const float*)d_in[p++];
    L[i].cwo  = (const float*)d_in[p++]; L[i].cong = (const float*)d_in[p++];
    L[i].cctx = (const float*)d_in[p++];
    L[i].hng  = (const float*)d_in[p++]; L[i].hwqkv= (const float*)d_in[p++];
    L[i].hwo  = (const float*)d_in[p++]; L[i].hong = (const float*)d_in[p++];
    L[i].fng  = (const float*)d_in[p++]; L[i].fw1  = (const float*)d_in[p++];
    L[i].fw2  = (const float*)d_in[p++];
  }
  const float* rel  = (const float*)d_in[p++];
  const float* outg = (const float*)d_in[p++];

  const int dinS[6]  = {768, 512, 512, 512, 512, 512};
  const int chout[6] = {512, 512, 512, 512, 512, 768};
  const int ffd[6]   = {512, 512, 512, 512, 512, 768};

  // workspace arena (all chunks 256B aligned)
  char* base = (char*)d_ws;
  size_t off = 0;
  auto allocf = [&](size_t n) { float* q = (float*)(base + off); off += ((n * 4 + 255) & ~(size_t)255); return q; };
  auto alloch = [&](size_t n) { unsigned short* q = (unsigned short*)(base + off); off += ((n * 2 + 255) & ~(size_t)255); return q; };

  float* BIAS = allocf(8ull * N_ * NP1_);          // rel-pos bias fp32
  float* X    = allocf(4096ull * 768);             // current x (fp32)
  float* TMP  = allocf(4096ull * 768);             // GEMM fp32 outputs
  float* RES  = allocf(4096ull * 768);             // branch output fp32
  float* KVR  = allocf(4ull * 1024 * 128);         // kv projection fp32
  float* QF   = allocf(32ull * 1024 * 64);         // attn out fp32 (b,h,n,d)
  float* SIM  = allocf(32ull * 1024 * KPAD_);      // scores / FF hidden / qkv fp32
  unsigned short* XN   = alloch(4096ull * 768);    // normalized (bf16)
  unsigned short* CTXN = alloch(4096ull * 256);
  unsigned short* ATT  = alloch(4096ull * 512);
  unsigned short* Q16  = alloch(2097152);          // (b,h,n,d) or (b,h,d,n)
  unsigned short* KT16 = alloch(2097152);          // chan K^T (b,h,d,n)
  unsigned short* VV16 = alloch(2097152);          // chan V   (b,h,n,d)
  unsigned short* KF16 = alloch(4ull * NKPAD_ * 64);
  unsigned short* VT16 = alloch(4ull * 64 * KPAD_);
  unsigned short* P16  = alloch(32ull * 1024 * KPAD_); // probs bf16 (also chan S)
  unsigned short* GT16 = alloch(4096ull * 3072);   // silu-gated bf16
  unsigned short* WT   = alloch(4718592);          // transposed weight bf16 (max)

  const long long rows  = 4096;                    // B*N
  const long long bhnd  = rows * 512;
  const long long kTot  = 4ll * NKPAD_ * 64;
  const long long vtTot = 4ll * 64 * KPAD_;
  const long long simS  = 1024ll * KPAD_;          // per (b,h) score stride

  k_relbias<<<cdiv((long long)N_ * NP1_, 256), 256, 0, stream>>>(rel, BIAS);

  for (int li = 0; li < 6; ++li) {
    const LP& P = L[li];
    const float* xcur = (li == 0) ? x_in : X;
    const int dS = dinS[li];

    // ---------------- self attention (causal, rotary, rel-pos bias) --------
    k_ln<unsigned short><<<rows, 256, 0, stream>>>(xcur, P.sng, XN, dS, 0);
    castT(stream, P.swq, WT, dS, 512);
    gemm(stream, XN, WT, TMP, 4096, 512, dS, dS, dS, 512, 0, 0, 0, 1, 1);
    k_to_bhnd<<<cdiv(bhnd, 256), 256, 0, stream>>>(TMP, 512, 0, Q16, 0.125f, 0, bhnd);
    castT(stream, P.swkv, WT, dS, 128);
    gemm(stream, XN, WT, KVR, 4096, 128, dS, dS, dS, 128, 0, 0, 0, 1, 1);
    k_build_k <<<cdiv(kTot, 256), 256, 0, stream>>>(KVR, P.snkv, KF16, kTot);
    k_build_vt<<<cdiv(vtTot, 256), 256, 0, stream>>>(KVR, P.snkv, VT16, vtTot);
    k_rotary<<<cdiv(bhnd / 4, 256), 256, 0, stream>>>(Q16, N_, 0, bhnd / 4);
    k_rotary<<<cdiv(4ll * NKPAD_ * 16, 256), 256, 0, stream>>>(KF16, NKPAD_, -1, 4ll * NKPAD_ * 16);
    gemm(stream, Q16, KF16, SIM, 1024, NP1_, 64, 64, 64, KPAD_,
         65536, (long long)NKPAD_ * 64, simS, 8, 32);
    k_softmax<<<32ll * 1024, 128, 0, stream>>>(SIM, BIAS, P16, 1024, 8, NP1_, KPAD_, 1);
    gemm(stream, P16, VT16, QF, 1024, 64, KPAD_, KPAD_, KPAD_, 64,
         simS, 64ll * KPAD_, 65536, 8, 32);
    k_to_bnhd<<<cdiv(bhnd, 256), 256, 0, stream>>>(QF, ATT, bhnd);
    castT(stream, P.swo, WT, 512, 512);
    gemm(stream, ATT, WT, TMP, 4096, 512, 512, 512, 512, 512, 0, 0, 0, 1, 1);
    k_ln<float><<<rows, 256, 0, stream>>>(TMP, P.song, RES, 512, 0);
    if (li == 0) k_copy<<<cdiv(rows * 512, 256), 256, 0, stream>>>(X, RES, rows * 512);
    else         k_add <<<cdiv(rows * 512, 256), 256, 0, stream>>>(X, RES, X, rows * 512);

    // ---------------- cross attention ---------------------------------------
    k_ln<unsigned short><<<rows, 256, 0, stream>>>(X, P.cng, XN, 512, 0);
    k_ln<unsigned short><<<rows, 256, 0, stream>>>(ctx, P.cctx, CTXN, 256, 0);
    castT(stream, P.cwq, WT, 512, 512);
    gemm(stream, XN, WT, TMP, 4096, 512, 512, 512, 512, 512, 0, 0, 0, 1, 1);
    k_to_bhnd<<<cdiv(bhnd, 256), 256, 0, stream>>>(TMP, 512, 0, Q16, 0.125f, 0, bhnd);
    castT(stream, P.cwkv, WT, 256, 128);
    gemm(stream, CTXN, WT, KVR, 4096, 128, 256, 256, 256, 128, 0, 0, 0, 1, 1);
    k_build_k <<<cdiv(kTot, 256), 256, 0, stream>>>(KVR, P.cnkv, KF16, kTot);
    k_build_vt<<<cdiv(vtTot, 256), 256, 0, stream>>>(KVR, P.cnkv, VT16, vtTot);
    gemm(stream, Q16, KF16, SIM, 1024, NP1_, 64, 64, 64, KPAD_,
         65536, (long long)NKPAD_ * 64, simS, 8, 32);
    k_softmax<<<32ll * 1024, 128, 0, stream>>>(SIM, nullptr, P16, 1024, 8, NP1_, KPAD_, 0);
    gemm(stream, P16, VT16, QF, 1024, 64, KPAD_, KPAD_, KPAD_, 64,
         simS, 64ll * KPAD_, 65536, 8, 32);
    k_to_bnhd<<<cdiv(bhnd, 256), 256, 0, stream>>>(QF, ATT, bhnd);
    castT(stream, P.cwo, WT, 512, 512);
    gemm(stream, ATT, WT, TMP, 4096, 512, 512, 512, 512, 512, 0, 0, 0, 1, 1);
    k_ln<float><<<rows, 256, 0, stream>>>(TMP, P.cong, RES, 512, 0);
    k_add<<<cdiv(rows * 512, 256), 256, 0, stream>>>(X, RES, X, rows * 512);

    // ---------------- channel attention -------------------------------------
    const int co = chout[li];
    k_ln<unsigned short><<<rows, 256, 0, stream>>>(X, P.hng, XN, 512, 0);
    castT(stream, P.hwqkv, WT, 512, 1536);
    gemm(stream, XN, WT, SIM, 4096, 1536, 512, 512, 512, 1536, 0, 0, 0, 1, 1);
    k_to_bhnd<<<cdiv(bhnd, 256), 256, 0, stream>>>(SIM, 1536, 0,    Q16, 0.125f, 1, bhnd); // Q^T
    k_to_bhnd<<<cdiv(bhnd, 256), 256, 0, stream>>>(SIM, 1536, 512,  KT16, 1.f,   1, bhnd); // K^T
    k_to_bhnd<<<cdiv(bhnd, 256), 256, 0, stream>>>(SIM, 1536, 1024, VV16, 1.f,   0, bhnd); // V
    gemm(stream, Q16, KT16, SIM, 64, 64, 1024, 1024, 1024, 64,
         65536, 65536, 4096, 1, 32);                         // S = Q^T K
    k_softmax<<<32ll * 64, 128, 0, stream>>>(SIM, nullptr, P16, 64, 8, 64, 64, 0);
    gemm(stream, VV16, P16, QF, 1024, 64, 64, 64, 64, 64,
         65536, 4096, 65536, 1, 32);                         // out = V S^T
    k_to_bnhd<<<cdiv(bhnd, 256), 256, 0, stream>>>(QF, ATT, bhnd);
    castT(stream, P.hwo, WT, 512, co);
    gemm(stream, ATT, WT, TMP, 4096, co, 512, 512, 512, co, 0, 0, 0, 1, 1);
    k_ln<float><<<rows, 256, 0, stream>>>(TMP, P.hong, RES, co, 0);
    if (li == 5) k_copy<<<cdiv(rows * co, 256), 256, 0, stream>>>(X, RES, rows * co);
    else         k_add <<<cdiv(rows * co, 256), 256, 0, stream>>>(X, RES, X, rows * co);

    // ---------------- feed-forward (SiLU gated) -----------------------------
    const int fd = ffd[li], inner = 4 * fd;
    k_ln<unsigned short><<<rows, 256, 0, stream>>>(X, P.fng, XN, fd, 0);
    castT(stream, P.fw1, WT, fd, 2 * inner);
    gemm(stream, XN, WT, SIM, 4096, 2 * inner, fd, fd, fd, 2 * inner, 0, 0, 0, 1, 1);
    k_silu<<<cdiv(rows * inner, 256), 256, 0, stream>>>(SIM, GT16, inner, rows * inner);
    castT(stream, P.fw2, WT, inner, fd);
    gemm(stream, GT16, WT, TMP, 4096, fd, inner, inner, inner, fd, 0, 0, 0, 1, 1);
    k_add<<<cdiv(rows * fd, 256), 256, 0, stream>>>(X, TMP, X, rows * fd);
  }

  k_ln<float><<<rows, 256, 0, stream>>>(X, outg, (float*)d_out, 768, 1);
}
